// CompGCNLayer_26190710571677
// MI455X (gfx1250) — compile-verified, matrix operand order checked by reading, not run on hardware
//
#include <hip/hip_runtime.h>
#include <hip/hip_bf16.h>

#define D 256
#define TE 16

typedef __attribute__((ext_vector_type(16))) __bf16 v16bf;
typedef __attribute__((ext_vector_type(8)))  float  v8f;

union Frag {
    unsigned int u[8];
    uint4 q[2];
    v16bf v;
};

static __device__ __forceinline__ unsigned f32_to_bf16u(float f) {
    unsigned u = __float_as_uint(f);
    return (u + 0x7FFFu + ((u >> 16) & 1u)) >> 16;  // RNE
}
static __device__ __forceinline__ unsigned pack_bf16(float lo, float hi) {
    return f32_to_bf16u(lo) | (f32_to_bf16u(hi) << 16);
}

// ---------------- zero workspace ----------------
__global__ void k_zero(float* __restrict__ agg, float* __restrict__ stats, long long n) {
    long long gid = (long long)blockIdx.x * blockDim.x + threadIdx.x;
    long long stride = (long long)gridDim.x * blockDim.x;
    for (long long i = gid; i < n; i += stride) agg[i] = 0.0f;
    if (gid < 1024) stats[gid] = 0.0f;
}

// ---------------- W_eff[j,o] = sum_k r[(j+k)%256] * w_loop[k,o] ----------------
__global__ void k_build_weff(const float* __restrict__ loop_rel,
                             const float* __restrict__ w_loop,
                             float* __restrict__ weff) {
    __shared__ float rS[D];
    int j = blockIdx.x, o = threadIdx.x;
    rS[o] = loop_rel[o];
    __syncthreads();
    float acc = 0.f;
    for (int k = 0; k < D; ++k)
        acc = fmaf(rS[(j + k) & 255], w_loop[k * D + o], acc);
    weff[j * D + o] = acc;
}

// ---------------- pack f32 [K=256][N=256] weight into bf16 WMMA B fragments --------
// layout: chunk = colTile*8 + kb ; per chunk [32 lanes][8 u32]
__global__ void k_pack_b(const float* __restrict__ W, unsigned* __restrict__ out) {
    int chunk = blockIdx.x;            // 0..127
    int c = chunk >> 3, kb = chunk & 7;
    int t = threadIdx.x;               // 0..255
    int l = t & 31, v = t >> 5;        // lane, vgpr-slot
    int n  = c * 16 + (l & 15);
    int hi = (l >= 16) ? 8 : 0;
    int kOff = (v < 4) ? (2 * v) : (16 + 2 * (v - 4));
    int k = kb * 32 + kOff + hi;
    out[(size_t)chunk * 256 + l * 8 + v] = pack_bf16(W[k * D + n], W[(k + 1) * D + n]);
}

// ---------------- self-loop GEMM: H = bf16(X) @ Bp + bias ----------------
__global__ __launch_bounds__(256) void k_loop_gemm(
    const float* __restrict__ X, const unsigned* __restrict__ Bp,
    const float* __restrict__ bias, float* __restrict__ H, int N) {
    __shared__ __align__(16) unsigned aS[TE * (D / 2)];   // 16 rows x 128 bf16-pairs
    int tid = threadIdx.x;
    int r0 = blockIdx.x * TE;
    for (int i = tid; i < TE * (D / 2); i += 256) {
        int m = i >> 7, ku = i & 127;
        int row = r0 + m; if (row >= N) row = N - 1;
        float2 f2 = *(const float2*)(X + (size_t)row * D + 2 * ku);
        aS[i] = pack_bf16(f2.x, f2.y);
    }
    __syncthreads();

    int lane = tid & 31, w = tid >> 5;
    int m = lane & 15;
    int hiU = (lane >= 16) ? 4 : 0;   // +8 k-offset in u32 units
    int c0 = w * 2;
    v8f acc0 = {}; v8f acc1 = {};
    for (int kb = 0; kb < 8; ++kb) {
        Frag a, b0, b1;
        #pragma unroll
        for (int v = 0; v < 8; ++v) {
            int kU = kb * 16 + ((v < 4) ? v : (8 + v - 4)) + hiU;
            a.u[v] = aS[m * 128 + kU];
        }
        const uint4* p0 = (const uint4*)(Bp + ((size_t)(c0 * 8 + kb) * 256 + lane * 8));
        b0.q[0] = p0[0]; b0.q[1] = p0[1];
        const uint4* p1 = (const uint4*)(Bp + ((size_t)((c0 + 1) * 8 + kb) * 256 + lane * 8));
        b1.q[0] = p1[0]; b1.q[1] = p1[1];
        acc0 = __builtin_amdgcn_wmma_f32_16x16x32_bf16(false, a.v, false, b0.v, (short)0, acc0, false, false);
        acc1 = __builtin_amdgcn_wmma_f32_16x16x32_bf16(false, a.v, false, b1.v, (short)0, acc1, false, false);
    }
    int nl = lane & 15;
    int mBase = (lane >= 16) ? 8 : 0;
    #pragma unroll
    for (int r = 0; r < 8; ++r) {
        int row = r0 + mBase + r;
        if (row < N) {
            int n0 = c0 * 16 + nl, n1 = n0 + 16;
            H[(size_t)row * D + n0] = acc0[r] + bias[n0];
            H[(size_t)row * D + n1] = acc1[r] + bias[n1];
        }
    }
}

// ---------------- fused edge kernel: gather -> ccorr -> bf16 -> WMMA -> atomic scatter
__global__ __launch_bounds__(256) void k_edge(
    const float* __restrict__ node_ft, const float* __restrict__ edge_ft,
    const int* __restrict__ src, const int* __restrict__ dst,
    const unsigned* __restrict__ Bp, float* __restrict__ agg, int E) {
    __shared__ __align__(16) float sF[TE][D];        // 16 KB gathered src rows
    __shared__ __align__(16) float fF[TE][2 * D];    // 32 KB edge rows, duplicated
    __shared__ __align__(16) unsigned ccB[TE * (D / 2)];  // 8 KB bf16 ccorr tile
    __shared__ int srcS[TE];
    __shared__ int dstS[TE];
    int tid = threadIdx.x;
    int e0 = blockIdx.x * TE;
    if (tid < TE) {
        int eg = e0 + tid;
        int egc = (eg < E) ? eg : (E - 1);
        srcS[tid] = src[egc];
        dstS[tid] = (eg < E) ? dst[eg] : -1;
    }
    __syncthreads();
    for (int i = tid; i < TE * D; i += 256) {           // gather s rows (coalesced per row)
        int e = i >> 8, j = i & 255;
        sF[e][j] = node_ft[(size_t)srcS[e] * D + j];
    }
    for (int i = tid; i < TE * 2 * D; i += 256) {       // edge rows, duplicated for wrap-free reads
        int e = i >> 9, j = i & 511;
        int eg = e0 + e; if (eg >= E) eg = E - 1;
        fF[e][j] = edge_ft[(size_t)eg * D + (j & 255)];
    }
    __syncthreads();

    // ccorr: cc[e,k] = sum_j s[j] * f[j+k]; thread -> (edge, 16 consecutive k)
    {
        int e = tid >> 4;
        int kbase = (tid & 15) << 4;
        float c[16];
        #pragma unroll
        for (int i = 0; i < 16; ++i) c[i] = 0.f;
        const float* sRow = &sF[e][0];
        const float* fRow = &fF[e][kbase];
        for (int j = 0; j < D; j += 4) {
            float4 s4 = *(const float4*)(sRow + j);
            float fv[20];
            #pragma unroll
            for (int q = 0; q < 5; ++q) {
                float4 t = *(const float4*)(fRow + j + 4 * q);
                fv[4 * q] = t.x; fv[4 * q + 1] = t.y; fv[4 * q + 2] = t.z; fv[4 * q + 3] = t.w;
            }
            #pragma unroll
            for (int kk = 0; kk < 16; ++kk) {
                c[kk] = fmaf(s4.x, fv[kk],     c[kk]);
                c[kk] = fmaf(s4.y, fv[kk + 1], c[kk]);
                c[kk] = fmaf(s4.z, fv[kk + 2], c[kk]);
                c[kk] = fmaf(s4.w, fv[kk + 3], c[kk]);
            }
        }
        unsigned* outRow = &ccB[e * 128 + (kbase >> 1)];
        #pragma unroll
        for (int i = 0; i < 8; ++i) outRow[i] = pack_bf16(c[2 * i], c[2 * i + 1]);
    }
    __syncthreads();

    // msg = cc @ w_in via WMMA, then scatter-add into agg[dst]
    int lane = tid & 31, w = tid >> 5;
    int m = lane & 15;
    int hiU = (lane >= 16) ? 4 : 0;
    int c0 = w * 2;
    v8f acc0 = {}; v8f acc1 = {};
    for (int kb = 0; kb < 8; ++kb) {
        Frag a, b0, b1;
        #pragma unroll
        for (int v = 0; v < 8; ++v) {
            int kU = kb * 16 + ((v < 4) ? v : (8 + v - 4)) + hiU;
            a.u[v] = ccB[m * 128 + kU];
        }
        const uint4* p0 = (const uint4*)(Bp + ((size_t)(c0 * 8 + kb) * 256 + lane * 8));
        b0.q[0] = p0[0]; b0.q[1] = p0[1];
        const uint4* p1 = (const uint4*)(Bp + ((size_t)((c0 + 1) * 8 + kb) * 256 + lane * 8));
        b1.q[0] = p1[0]; b1.q[1] = p1[1];
        acc0 = __builtin_amdgcn_wmma_f32_16x16x32_bf16(false, a.v, false, b0.v, (short)0, acc0, false, false);
        acc1 = __builtin_amdgcn_wmma_f32_16x16x32_bf16(false, a.v, false, b1.v, (short)0, acc1, false, false);
    }
    int nl = lane & 15;
    int mBase = (lane >= 16) ? 8 : 0;
    #pragma unroll
    for (int r = 0; r < 8; ++r) {
        int e = mBase + r;
        int dn = dstS[e];
        if (dn >= 0) {
            atomicAdd(&agg[(size_t)dn * D + c0 * 16 + nl], acc0[r]);
            atomicAdd(&agg[(size_t)dn * D + (c0 + 1) * 16 + nl], acc1[r]);
        }
    }
}

// ---------------- BN stats: fold agg*norm into H, accumulate per-feature sums ----
__global__ __launch_bounds__(256) void k_bn_stats(
    float* __restrict__ H, const float* __restrict__ agg,
    const float* __restrict__ normv, float* __restrict__ stats, int N) {
    int o = threadIdx.x;
    int n0 = blockIdx.x * 128;
    int n1 = n0 + 128; if (n1 > N) n1 = N;
    float s = 0.f, sq = 0.f;
    for (int n = n0; n < n1; ++n) {
        float t = fmaf(agg[(size_t)n * D + o], normv[n], H[(size_t)n * D + o]);
        H[(size_t)n * D + o] = t;
        s += t;
        sq = fmaf(t, t, sq);
    }
    atomicAdd(&stats[o], s);
    atomicAdd(&stats[D + o], sq);
}

__global__ void k_bn_final(float* __restrict__ stats, const float* __restrict__ gamma,
                           const float* __restrict__ beta, int N) {
    int o = threadIdx.x;
    float inv = 1.0f / (float)N;
    float mean = stats[o] * inv;
    float var = stats[D + o] * inv - mean * mean;
    float sc = gamma[o] * rsqrtf(var + 1e-5f);
    stats[2 * D + o] = sc;
    stats[3 * D + o] = beta[o] - mean * sc;
}

__global__ __launch_bounds__(256) void k_bn_apply(
    const float* __restrict__ H, const float* __restrict__ stats,
    float* __restrict__ out, long long total) {
    long long i4 = ((long long)blockIdx.x * 256 + threadIdx.x) * 4;
    if (i4 >= total) return;
    int o = (int)(i4 & 255);
    float4 h = *(const float4*)(H + i4);
    float4 sc = *(const float4*)(stats + 2 * D + o);
    float4 sh = *(const float4*)(stats + 3 * D + o);
    float4 r;
    r.x = tanhf(fmaf(h.x, sc.x, sh.x));
    r.y = tanhf(fmaf(h.y, sc.y, sh.y));
    r.z = tanhf(fmaf(h.z, sc.z, sh.z));
    r.w = tanhf(fmaf(h.w, sc.w, sh.w));
    *(float4*)(out + i4) = r;
}

extern "C" void kernel_launch(void* const* d_in, const int* in_sizes, int n_in,
                              void* d_out, int out_size, void* d_ws, size_t ws_size,
                              hipStream_t stream) {
    const float* node_ft  = (const float*)d_in[0];
    const float* edge_ft  = (const float*)d_in[1];
    const float* normv    = (const float*)d_in[2];
    const float* w_loop   = (const float*)d_in[3];
    const float* w_in     = (const float*)d_in[4];
    const float* loop_rel = (const float*)d_in[5];
    const float* w_bias   = (const float*)d_in[6];
    const float* gamma    = (const float*)d_in[7];
    const float* beta     = (const float*)d_in[8];
    const int*   src      = (const int*)d_in[9];
    const int*   dst      = (const int*)d_in[10];
    int N = in_sizes[2];   // norm is (N,1)
    int E = in_sizes[9];   // src is (E,)

    // workspace layout
    float*    AGG      = (float*)d_ws;                      // N*256 f32
    float*    H        = AGG + (size_t)N * D;               // N*256 f32
    float*    WeffF    = H + (size_t)N * D;                 // 256*256 f32
    unsigned* packWin  = (unsigned*)(WeffF + D * D);        // 32768 u32
    unsigned* packWeff = packWin + 16 * 8 * 256;            // 32768 u32
    float*    stats    = (float*)(packWeff + 16 * 8 * 256); // 1024 f32

    long long nAgg = (long long)N * D;
    k_zero<<<2048, 256, 0, stream>>>(AGG, stats, nAgg);
    k_build_weff<<<D, D, 0, stream>>>(loop_rel, w_loop, WeffF);
    k_pack_b<<<128, 256, 0, stream>>>(WeffF, packWeff);
    k_pack_b<<<128, 256, 0, stream>>>(w_in, packWin);
    k_loop_gemm<<<(N + TE - 1) / TE, 256, 0, stream>>>(node_ft, packWeff, w_bias, H, N);
    k_edge<<<(E + TE - 1) / TE, 256, 0, stream>>>(node_ft, edge_ft, src, dst, packWin, AGG, E);
    k_bn_stats<<<(N + 127) / 128, 256, 0, stream>>>(H, AGG, normv, stats, N);
    k_bn_final<<<1, 256, 0, stream>>>(stats, gamma, beta, N);
    long long total = (long long)N * D;
    k_bn_apply<<<(int)((total / 4 + 255) / 256), 256, 0, stream>>>(H, stats, (float*)d_out, total);
}
